// GroupedQueryAttention_11295763988511
// MI455X (gfx1250) — compile-verified, matrix operand order checked by reading, not run on hardware
//
#include <hip/hip_runtime.h>
#include <hip/hip_bf16.h>
#include <hip/hip_fp16.h>

// ---------------- problem constants ----------------
#define B_   2
#define T_   2048
#define C_   2048
#define H_   16
#define HD_  128
#define G_   4          // KV groups
#define KV_  512        // G_*HD_
#define M_   (B_*T_)    // 4096 rows

typedef _Float16 f16;
typedef __attribute__((ext_vector_type(4)))  _Float16 v4h;
typedef __attribute__((ext_vector_type(8)))  _Float16 v8h;
typedef __attribute__((ext_vector_type(16))) _Float16 v16h;
typedef __attribute__((ext_vector_type(8)))  float    v8f;

// Load one 16x32 f16 WMMA fragment half-row: 8 contiguous f16 at p, 8 at p+16
// (16-bit A/B VGPR layout: elems 0..7 -> K=kbase..+7, elems 8..15 -> K=kbase+16..+23,
//  kbase = (lane>>4)*8 folded into p).
static __device__ __forceinline__ v16h ldfrag(const f16* p) {
  v8h lo = *(const v8h*)(p);
  v8h hi = *(const v8h*)(p + 16);
  return __builtin_shufflevector(lo, hi, 0,1,2,3,4,5,6,7,8,9,10,11,12,13,14,15);
}

static __device__ __forceinline__ v8f wmma16(v16h a, v16h b, v8f c) {
  return __builtin_amdgcn_wmma_f32_16x16x32_f16(false, a, false, b, (short)0, c,
                                                false, false);
}

// ---------------- elementwise conversion ----------------
__global__ void cvt_f32_f16(const float* __restrict__ src,
                            f16* __restrict__ dst, int n) {
  int i = blockIdx.x * blockDim.x + threadIdx.x;
  if (i < n) dst[i] = (f16)src[i];
}

// Convert + transpose weights: W (K x N, f32) -> Wt (N x K, f16)
__global__ void wtrans_f16(const float* __restrict__ W,
                           f16* __restrict__ Wt, int K, int N) {
  int i = blockIdx.x * blockDim.x + threadIdx.x;
  if (i >= K * N) return;
  int k = i / N, n = i - k * N;
  Wt[(size_t)n * K + k] = (f16)W[i];
}

// ---------------- WMMA GEMM with async-LDS staged B tile ----------------
// C(MxN,f32) = A(MxK,f16) @ Bt(NxK,f16)^T + bias.
// Block = 4 waves = 64 M-rows x 64 N-cols. The 64x32 B k-slice (4 KB) is
// shared by all 4 waves: staged global->LDS with GLOBAL_LOAD_ASYNC_TO_LDS_B128
// (ASYNCcnt), double-buffered (8 KB dynamic LDS), one barrier per stage, last
// stage peeled so the steady-state loop is branch-free.
// A fragments stream straight from L2 (192 MB L2 holds the whole working set).
__global__ void __launch_bounds__(128)
gemm_wmma(const f16* __restrict__ A, const f16* __restrict__ Bt,
          const float* __restrict__ bias, float* __restrict__ Cout,
          int M, int N, int K) {
  extern __shared__ __align__(16) f16 smem[];   // 2 * 64*32 f16 = 8192 bytes
  const int tid  = threadIdx.x;
  const int wave = tid >> 5;
  const int lane = tid & 31;
  const int lh = lane & 15, ks = lane >> 4;
  const int mrow = (blockIdx.x * 4 + wave) * 16 + lh;
  const int n0   = blockIdx.y * 64;
  const f16* arow = A + (size_t)mrow * K;

  // Stage the 64x32 B tile for k-slice k0 into buffer `buf`.
  // 4 KB = 256 16-byte chunks; each of the 128 threads moves 2 chunks.
  // chunk c: row = c>>2 (64B per row), koff = (c&3)*8 f16.
  auto issue_stage = [&](int k0, int buf) {
    #pragma unroll
    for (int u = 0; u < 2; ++u) {
      const int c   = tid + u * 128;
      const int row = c >> 2;
      const int ko  = (c & 3) * 8;
      const f16* g  = Bt + (size_t)(n0 + row) * K + k0 + ko;
      const unsigned lds = (unsigned)(buf * 4096 + (row * 32 + ko) * 2);
      asm volatile("global_load_async_to_lds_b128 %0, %1, off"
                   :: "v"(lds), "v"(g) : "memory");
    }
  };

  v8f acc[4] = {};

  // One k-step: load A frag (global) + all 4 B frags (LDS) into distinct
  // values first, then issue the 4 WMMAs -> lets the compiler clause the
  // ds_loads and use partial DScnt waits instead of a full drain per WMMA.
  auto kstep = [&](int k0, int buf) {
    __builtin_prefetch(arow + k0 + 512, 0, 3);       // pull A ahead into near $
    const v16h af  = ldfrag(arow + k0 + ks * 8);
    const f16* sb  = smem + buf * 2048 + lh * 32 + ks * 8;
    const v16h bf0 = ldfrag(sb + 0 * 512);
    const v16h bf1 = ldfrag(sb + 1 * 512);
    const v16h bf2 = ldfrag(sb + 2 * 512);
    const v16h bf3 = ldfrag(sb + 3 * 512);
    acc[0] = wmma16(af, bf0, acc[0]);
    acc[1] = wmma16(af, bf1, acc[1]);
    acc[2] = wmma16(af, bf2, acc[2]);
    acc[3] = wmma16(af, bf3, acc[3]);
  };

  issue_stage(0, 0);
  int cur = 0;
  for (int k0 = 0; k0 < K - 32; k0 += 32) {
    // my portion of buffer `cur` has landed in LDS
    asm volatile("s_wait_asynccnt 0x0" ::: "memory");
    // everyone's portion landed; everyone done reading buffer cur^1
    __syncthreads();
    issue_stage(k0 + 32, cur ^ 1);                    // overlap next DMA
    kstep(k0, cur);
    cur ^= 1;
  }
  // peeled last stage (no further DMA to issue)
  asm volatile("s_wait_asynccnt 0x0" ::: "memory");
  __syncthreads();
  kstep(K - 32, cur);

  // C/D layout: lane<16 -> rows 0..7, lane>=16 -> rows 8..15; col = n-tile + lh
  const int rbase = (blockIdx.x * 4 + wave) * 16 + ks * 8;
  #pragma unroll
  for (int j = 0; j < 4; ++j) {
    const int col = n0 + j * 16 + lh;
    const float bj = bias[col];
    #pragma unroll
    for (int r = 0; r < 8; ++r)
      Cout[(size_t)(rbase + r) * N + col] = acc[j][r] + bj;
  }
}

// ---------------- L2-normalize (+alpha) + RoPE, f32 -> f16 ----------------
// One wave per (row=b*T+t, head). Lane owns 4 contiguous d's (two RoPE pairs).
// Reference multiplies Q by alpha*sqrt(hd) then divides scores by sqrt(hd):
// folded -> multiply by alpha here; attention uses the raw dot product.
__global__ void __launch_bounds__(32)
normrope(const float* __restrict__ src, f16* __restrict__ dst,
         const float* __restrict__ alpha, int stride, int useAlpha) {
  const int row  = blockIdx.x;       // b*T + t
  const int h    = blockIdx.y;
  const int t    = row & (T_ - 1);
  const int lane = threadIdx.x;
  const float* p = src + (size_t)row * stride + h * HD_ + lane * 4;
  float4 v = *(const float4*)p;
  float ss = v.x*v.x + v.y*v.y + v.z*v.z + v.w*v.w;
  #pragma unroll
  for (int m = 16; m >= 1; m >>= 1) ss += __shfl_xor(ss, m);
  float sc = 1.0f / (sqrtf(ss) + 1e-6f);
  if (useAlpha) sc *= alpha[h];

  const float LOG10000 = 9.210340371976184f;
  float ia = (float)(2 * lane)     * (1.0f / 64.0f);   // pair index / 64
  float ib = (float)(2 * lane + 1) * (1.0f / 64.0f);
  float fa = (float)t * __expf(-ia * LOG10000);
  float fb = (float)t * __expf(-ib * LOG10000);
  float ca = __cosf(fa), sa = __sinf(fa);
  float cb = __cosf(fb), sb = __sinf(fb);
  float x0 = v.x * sc, x1 = v.y * sc, x2 = v.z * sc, x3 = v.w * sc;
  v4h out;
  out[0] = (f16)(x0 * ca - x1 * sa);
  out[1] = (f16)(x0 * sa + x1 * ca);
  out[2] = (f16)(x2 * cb - x3 * sb);
  out[3] = (f16)(x2 * sb + x3 * cb);
  *(v4h*)(dst + (size_t)row * stride + h * HD_ + lane * 4) = out;
}

// ---------------- V transpose: (B*T, KV) f32 -> (B, G, HD, T) f16 ----------------
__global__ void vtrans(const float* __restrict__ v32, f16* __restrict__ vT) {
  int i = blockIdx.x * blockDim.x + threadIdx.x;
  if (i >= M_ * KV_) return;
  int dkv = i & (KV_ - 1);
  int row = i >> 9;            // / KV_
  int t   = row & (T_ - 1);
  int b   = row >> 11;         // / T_
  int g   = dkv >> 7;          // / HD_
  int d   = dkv & (HD_ - 1);
  vT[((size_t)((b * G_ + g) * HD_ + d)) * T_ + t] = (f16)v32[i];
}

// ---------------- Flash attention (transposed-WMMA formulation) ----------------
// One wave per (16-query tile, head, batch). Key tiles of 32.
//   S^T = K(32x128) @ Q^T(128x16)  -> lane owns one query column (lane pair)
//   O^T = V^T(128x32) @ P^T(32x16) -> S^T C-layout == P^T B-layout (no shuffle!)
__global__ void __launch_bounds__(32)
flash_attn(const f16* __restrict__ q16, const f16* __restrict__ k16,
           const f16* __restrict__ vT16, f16* __restrict__ y16) {
  const int qt = blockIdx.x, h = blockIdx.y, b = blockIdx.z;
  const int g  = h >> 2;                    // jnp.repeat: head h -> group h/4
  const int lane = threadIdx.x;
  const int lh = lane & 15, ks = lane >> 4;
  const int qbase = qt * 16;
  const int qcol  = qbase + lh;             // this lane's query column

  // Q^T B-fragments: 4 fragments covering d = 0..127 (K=32 each)
  const f16* qrow = q16 + ((size_t)(b * T_ + qcol)) * C_ + h * HD_;
  v16h qf0 = ldfrag(qrow +   0 + ks * 8);
  v16h qf1 = ldfrag(qrow +  32 + ks * 8);
  v16h qf2 = ldfrag(qrow +  64 + ks * 8);
  v16h qf3 = ldfrag(qrow +  96 + ks * 8);

  const f16* kb = k16  + ((size_t)(b * T_)) * KV_ + g * HD_;
  const f16* vb = vT16 + ((size_t)(b * G_ + g)) * HD_ * T_;

  v8f o[8] = {};
  float mrow = -1e30f, lrow = 0.0f;

  const int ntiles = (qbase + 16 + 31) >> 5;  // causal: keys <= qbase+15
  for (int kt = 0; kt < ntiles; ++kt) {
    const int s0 = kt << 5;
    // S^T: two 16-key row groups, contraction over d in 4 steps of 32
    v8f st0 = {}, st1 = {};
    const f16* kr0 = kb + (size_t)(s0 + lh) * KV_ + ks * 8;
    const f16* kr1 = kr0 + (size_t)16 * KV_;
    {
      v16h a;
      a = ldfrag(kr0 +  0); st0 = wmma16(a, qf0, st0);
      a = ldfrag(kr0 + 32); st0 = wmma16(a, qf1, st0);
      a = ldfrag(kr0 + 64); st0 = wmma16(a, qf2, st0);
      a = ldfrag(kr0 + 96); st0 = wmma16(a, qf3, st0);
      a = ldfrag(kr1 +  0); st1 = wmma16(a, qf0, st1);
      a = ldfrag(kr1 + 32); st1 = wmma16(a, qf1, st1);
      a = ldfrag(kr1 + 64); st1 = wmma16(a, qf2, st1);
      a = ldfrag(kr1 + 96); st1 = wmma16(a, qf3, st1);
    }
    // Causal mask + online softmax. Lane holds keys {kofs..+7, kofs+16..+23}.
    const int kofs = s0 + ks * 8;
    float sv[16];
    #pragma unroll
    for (int r = 0; r < 8; ++r) {
      sv[r]     = (kofs + r      <= qcol) ? st0[r] : -1e30f;
      sv[r + 8] = (kofs + r + 16 <= qcol) ? st1[r] : -1e30f;
    }
    float tmax = sv[0];
    #pragma unroll
    for (int i = 1; i < 16; ++i) tmax = fmaxf(tmax, sv[i]);
    tmax = fmaxf(tmax, __shfl_xor(tmax, 16));     // other half of this query
    float mnew  = fmaxf(mrow, tmax);
    float scale = __expf(mrow - mnew);
    v16h pf;                                       // P^T B-fragment, in place
    float psum = 0.0f;
    #pragma unroll
    for (int i = 0; i < 16; ++i) {
      float pv = __expf(sv[i] - mnew);
      psum += pv;
      pf[i] = (f16)pv;
    }
    psum += __shfl_xor(psum, 16);
    lrow = lrow * scale + psum;
    mrow = mnew;
    #pragma unroll
    for (int dt = 0; dt < 8; ++dt) o[dt] = o[dt] * scale;
    // O^T += V^T @ P^T : 8 d-tiles of 16
    const f16* vr = vb + (size_t)lh * T_ + s0 + ks * 8;
    #pragma unroll
    for (int dt = 0; dt < 8; ++dt) {
      v16h a = ldfrag(vr + (size_t)dt * 16 * T_);
      o[dt] = wmma16(a, pf, o[dt]);
    }
  }

  // Normalize and write y (B,T,C) f16: lane owns query qcol, d = dt*16+ks*8+r
  const float inv = 1.0f / lrow;
  f16* yp = y16 + ((size_t)(b * T_ + qcol)) * C_ + h * HD_ + ks * 8;
  #pragma unroll
  for (int dt = 0; dt < 8; ++dt)
    #pragma unroll
    for (int r = 0; r < 8; ++r)
      yp[dt * 16 + r] = (f16)(o[dt][r] * inv);
}

// ---------------- host orchestration ----------------
extern "C" void kernel_launch(void* const* d_in, const int* in_sizes, int n_in,
                              void* d_out, int out_size, void* d_ws, size_t ws_size,
                              hipStream_t stream) {
  (void)in_sizes; (void)n_in; (void)out_size; (void)ws_size;
  const float* x     = (const float*)d_in[0];
  const float* Wq    = (const float*)d_in[1];
  const float* bq    = (const float*)d_in[2];
  const float* Wk    = (const float*)d_in[3];
  const float* bk    = (const float*)d_in[4];
  const float* Wv    = (const float*)d_in[5];
  const float* bv    = (const float*)d_in[6];
  const float* Wo    = (const float*)d_in[7];
  const float* bo    = (const float*)d_in[8];
  const float* alpha = (const float*)d_in[9];
  float* out = (float*)d_out;

  char* ws = (char*)d_ws;
  size_t off = 0;
  auto alloc = [&](size_t bytes) -> char* {
    char* p = ws + off;
    off += (bytes + 255) & ~(size_t)255;
    return p;
  };
  f16*   x16  = (f16*)alloc((size_t)M_ * C_  * 2);
  f16*   WqT  = (f16*)alloc((size_t)C_ * C_  * 2);
  f16*   WkT  = (f16*)alloc((size_t)C_ * KV_ * 2);
  f16*   WvT  = (f16*)alloc((size_t)C_ * KV_ * 2);
  f16*   WoT  = (f16*)alloc((size_t)C_ * C_  * 2);
  float* q32  = (float*)alloc((size_t)M_ * C_  * 4);
  float* k32  = (float*)alloc((size_t)M_ * KV_ * 4);
  float* v32  = (float*)alloc((size_t)M_ * KV_ * 4);
  f16*   q16  = (f16*)alloc((size_t)M_ * C_  * 2);
  f16*   k16  = (f16*)alloc((size_t)M_ * KV_ * 2);
  f16*   vT16 = (f16*)alloc((size_t)M_ * KV_ * 2);
  f16*   y16  = (f16*)alloc((size_t)M_ * C_  * 2);

  const int TB = 256;
  const int LDSB = 8192;   // double-buffered 64x32 f16 B tile

  // 1) precision conversion (x) and transpose-convert (weights)
  {
    int n = M_ * C_;
    cvt_f32_f16<<<(n + TB - 1) / TB, TB, 0, stream>>>(x, x16, n);
  }
  wtrans_f16<<<(C_ * C_  + TB - 1) / TB, TB, 0, stream>>>(Wq, WqT, C_, C_);
  wtrans_f16<<<(C_ * KV_ + TB - 1) / TB, TB, 0, stream>>>(Wk, WkT, C_, KV_);
  wtrans_f16<<<(C_ * KV_ + TB - 1) / TB, TB, 0, stream>>>(Wv, WvT, C_, KV_);
  wtrans_f16<<<(C_ * C_  + TB - 1) / TB, TB, 0, stream>>>(Wo, WoT, C_, C_);

  // 2) QKV projections (WMMA, async-LDS B tiles)
  gemm_wmma<<<dim3(M_ / 64, C_  / 64), 128, LDSB, stream>>>(x16, WqT, bq, q32, M_, C_,  C_);
  gemm_wmma<<<dim3(M_ / 64, KV_ / 64), 128, LDSB, stream>>>(x16, WkT, bk, k32, M_, KV_, C_);
  gemm_wmma<<<dim3(M_ / 64, KV_ / 64), 128, LDSB, stream>>>(x16, WvT, bv, v32, M_, KV_, C_);

  // 3) QK normalize (+alpha for Q; sqrt(hd) factors cancel) + RoPE -> f16
  normrope<<<dim3(M_, H_), 32, 0, stream>>>(q32, q16, alpha, C_,  1);
  normrope<<<dim3(M_, G_), 32, 0, stream>>>(k32, k16, alpha, KV_, 0);

  // 4) V -> V^T f16 (B,G,HD,T)
  {
    int n = M_ * KV_;
    vtrans<<<(n + TB - 1) / TB, TB, 0, stream>>>(v32, vT16);
  }

  // 5) causal flash attention (WMMA)
  flash_attn<<<dim3(T_ / 16, H_, B_), 32, 0, stream>>>(q16, k16, vT16, y16);

  // 6) output projection (WMMA) -> f32 d_out
  gemm_wmma<<<dim3(M_ / 64, C_ / 64), 128, LDSB, stream>>>(y16, WoT, bo, out, M_, C_, C_);
}
